// BasicTransformer_78288663872115
// MI455X (gfx1250) — compile-verified, matrix operand order checked by reading, not run on hardware
//
#include <hip/hip_runtime.h>

#define DEVFN __device__ __forceinline__

static constexpr int B_ = 2, S_ = 2048, H_ = 1024, NH_ = 16, FFN_ = 4096, D_ = 64;
static constexpr int BS_ = B_ * S_;

typedef __attribute__((ext_vector_type(16))) __bf16 v16bf;
typedef __attribute__((ext_vector_type(8)))  float  v8f;

union ABFrag { v16bf v; unsigned short h[16]; unsigned int d[8]; };
union CFrag  { v8f v; float f[8]; };
union U4     { uint4 q; unsigned int w[4]; unsigned short h[8]; };

DEVFN unsigned short f32_to_bf16(float f) {
  union { float f; unsigned int u; } c; c.f = f;
  unsigned int u = c.u + 0x7FFFu + ((c.u >> 16) & 1u);  // round-to-nearest-even
  return (unsigned short)(u >> 16);
}

// ---------------------------------------------------------------------------
// CDNA5 async global->LDS copy (16 bytes per lane, tracked by ASYNCcnt).
// Builtin signature (from clang diagnostic): param0 = AS(1) ptr to int
// vector_size(16), param1 = AS(3) counterpart, then (imm offset, imm cpol).
// ---------------------------------------------------------------------------
DEVFN void async_cp16(const unsigned short* g, unsigned short* l) {
#if __has_builtin(__builtin_amdgcn_global_load_async_to_lds_b128)
  typedef int b128 __attribute__((vector_size(16)));
  typedef __attribute__((address_space(1))) b128* gp_t;
  typedef __attribute__((address_space(3))) b128* lp_t;
  __builtin_amdgcn_global_load_async_to_lds_b128(
      (gp_t)(unsigned long long)g,
      (lp_t)(__attribute__((address_space(3))) void*)l, 0, 0);
#else
  unsigned int loff =
      (unsigned int)(unsigned long long)(__attribute__((address_space(3))) void*)l;
  asm volatile("global_load_async_to_lds_b128 %0, %1, off"
               :: "v"(loff), "v"(g) : "memory");
#endif
}

DEVFN void wait_async0() {
#if __has_builtin(__builtin_amdgcn_s_wait_asynccnt)
  __builtin_amdgcn_s_wait_asynccnt(0);
#else
  asm volatile("s_wait_asynccnt 0" ::: "memory");
#endif
}

// Pack a 16x32 bf16 A/B fragment from a row-major ushort matrix.
// Per CDNA5 ISA layout: lane<16 holds K in {0..7,16..23}, lane>=16 holds K in
// {8..15,24..31}; pairs (2i,2i+1) are contiguous -> 4-byte loads (compiler
// merges them into ds_load_b128).
DEVFN void load_frag16(ABFrag& fr, const unsigned short* base, int row, int stride, int klo) {
  const unsigned short* p = base + (size_t)row * stride;
  #pragma unroll
  for (int i = 0; i < 4; ++i) {
    fr.d[i]     = *(const unsigned int*)(p + klo + 2 * i);
    fr.d[i + 4] = *(const unsigned int*)(p + 16 + klo + 2 * i);
  }
}

DEVFN v8f wmma_bf16(const ABFrag& a, const ABFrag& b, v8f c) {
  return __builtin_amdgcn_wmma_f32_16x16x32_bf16(
      /*neg_a=*/false, a.v, /*neg_b=*/false, b.v,
      /*c_mod=*/(short)0, c, /*reuse_a=*/false, /*reuse_b=*/false);
}

// ---------------------------------------------------------------------------
// fp32 [K][N] -> bf16 [N][K] tiled transpose (weights), 32x32 tiles
// ---------------------------------------------------------------------------
__global__ __launch_bounds__(256) void cvt_transpose_bf16_kernel(
    const float* __restrict__ in, unsigned short* __restrict__ out, int K, int N) {
  __shared__ float tile[32][33];
  const int k0 = blockIdx.y * 32;
  const int n0 = blockIdx.x * 32;
  const int r = threadIdx.x >> 3;
  const int c = (threadIdx.x & 7) << 2;
  float4 v = *(const float4*)(in + (size_t)(k0 + r) * N + n0 + c);
  tile[r][c + 0] = v.x; tile[r][c + 1] = v.y;
  tile[r][c + 2] = v.z; tile[r][c + 3] = v.w;
  __syncthreads();
  unsigned short* op = out + (size_t)(n0 + r) * K + k0 + c;
  op[0] = f32_to_bf16(tile[c + 0][r]);
  op[1] = f32_to_bf16(tile[c + 1][r]);
  op[2] = f32_to_bf16(tile[c + 2][r]);
  op[3] = f32_to_bf16(tile[c + 3][r]);
}

// ---------------------------------------------------------------------------
// LayerNorm over H=1024, one block (256 threads) per row, bf16 output
// ---------------------------------------------------------------------------
__global__ __launch_bounds__(256) void ln_bf16_kernel(
    const float* __restrict__ x, const float* __restrict__ g,
    const float* __restrict__ b, unsigned short* __restrict__ out) {
  __shared__ float s1[256], s2[256];
  const int row = blockIdx.x, t = threadIdx.x;
  const float* xp = x + (size_t)row * H_;
  float4 v = *(const float4*)(xp + t * 4);
  s1[t] = v.x + v.y + v.z + v.w;
  s2[t] = v.x * v.x + v.y * v.y + v.z * v.z + v.w * v.w;
  __syncthreads();
  for (int st = 128; st > 0; st >>= 1) {
    if (t < st) { s1[t] += s1[t + st]; s2[t] += s2[t + st]; }
    __syncthreads();
  }
  const float mean = s1[0] * (1.0f / H_);
  const float var  = s2[0] * (1.0f / H_) - mean * mean;
  const float rs   = rsqrtf(var + 1e-3f);
  const int c = t * 4;
  unsigned short* op = out + (size_t)row * H_ + c;
  op[0] = f32_to_bf16((v.x - mean) * rs * g[c + 0] + b[c + 0]);
  op[1] = f32_to_bf16((v.y - mean) * rs * g[c + 1] + b[c + 1]);
  op[2] = f32_to_bf16((v.z - mean) * rs * g[c + 2] + b[c + 2]);
  op[3] = f32_to_bf16((v.w - mean) * rs * g[c + 3] + b[c + 3]);
}

// ---------------------------------------------------------------------------
// Tiled bf16 WMMA GEMM with async double-buffered LDS staging.
// C[M,N] = A[M,K] * Bt[N,K]^T (+bias). 128x128 tile, K-step 32, 8 waves (4x2),
// each wave 32x64 (2x4 frags). Tiles staged via GLOBAL_LOAD_ASYNC_TO_LDS_B128.
// MODE 0: QKV scatter -> bf16 q/k/v [B,NH,S,D]
// MODE 1: + residual -> fp32 out
// MODE 2: gelu(tanh) -> bf16 out
// ---------------------------------------------------------------------------
template <int MODE>
__global__ __launch_bounds__(256) void gemm_bf16_wmma(
    const unsigned short* __restrict__ A,
    const unsigned short* __restrict__ Bt,   // transposed weights [N][K]
    const float* __restrict__ bias,
    const float* __restrict__ res,
    float* __restrict__ outF,
    unsigned short* __restrict__ outH,
    unsigned short* __restrict__ qO,
    unsigned short* __restrict__ kO,
    unsigned short* __restrict__ vO,
    int M, int N, int K) {
  __shared__ unsigned short As[2][128][40];  // [buf][m][k]
  __shared__ unsigned short Bs[2][128][40];  // [buf][n][k]

  const int tid  = threadIdx.x;
  const int lane = tid & 31;
  const int wave = tid >> 5;
  const int rsel = lane & 15;
  const int klo  = (lane >> 4) << 3;
  const int rowoff = (lane >> 4) << 3;
  const int wm = wave >> 1;
  const int wn = wave & 1;
  const int blockN = blockIdx.x * 128;
  const int blockM = blockIdx.y * 128;

  CFrag acc[2][4];
  #pragma unroll
  for (int mf = 0; mf < 2; ++mf)
    #pragma unroll
    for (int nf = 0; nf < 4; ++nf)
      acc[mf][nf].v = (v8f){0.f, 0.f, 0.f, 0.f, 0.f, 0.f, 0.f, 0.f};

  // Each thread issues 2 A-chunks + 2 B-chunks of 16B per tile (8KB+8KB).
  auto stage = [&](int buf, int kt) {
    #pragma unroll
    for (int j = 0; j < 2; ++j) {
      const int c   = j * 256 + tid;   // 0..511
      const int row = c >> 2;          // 0..127
      const int off = (c & 3) << 3;    // 0,8,16,24 ushorts (16B chunks)
      async_cp16(A  + (size_t)(blockM + row) * K + kt + off, &As[buf][row][off]);
      async_cp16(Bt + (size_t)(blockN + row) * K + kt + off, &Bs[buf][row][off]);
    }
  };

  stage(0, 0);
  wait_async0();
  __syncthreads();

  int buf = 0;
  for (int kt = 0; kt < K; kt += 32) {
    const bool more = (kt + 32) < K;
    if (more) stage(buf ^ 1, kt + 32);  // overlap next-tile DMA with WMMAs

    ABFrag af[2], bf[4];
    #pragma unroll
    for (int mf = 0; mf < 2; ++mf)
      load_frag16(af[mf], &As[buf][0][0], wm * 32 + mf * 16 + rsel, 40, klo);
    #pragma unroll
    for (int nf = 0; nf < 4; ++nf)
      load_frag16(bf[nf], &Bs[buf][0][0], wn * 64 + nf * 16 + rsel, 40, klo);
    #pragma unroll
    for (int mf = 0; mf < 2; ++mf)
      #pragma unroll
      for (int nf = 0; nf < 4; ++nf)
        acc[mf][nf].v = wmma_bf16(af[mf], bf[nf], acc[mf][nf].v);

    if (more) wait_async0();
    __syncthreads();
    buf ^= 1;
  }

  // Epilogue. C layout: row = vgpr r + 8*(lane>=16), col = lane&15.
  #pragma unroll
  for (int mf = 0; mf < 2; ++mf) {
    #pragma unroll
    for (int nf = 0; nf < 4; ++nf) {
      const int gc = blockN + wn * 64 + nf * 16 + rsel;
      const float bia = bias[gc];
      #pragma unroll
      for (int r = 0; r < 8; ++r) {
        const int gr = blockM + wm * 32 + mf * 16 + rowoff + r;
        float val = acc[mf][nf].f[r] + bia;
        if (MODE == 0) {
          const int head  = gc / (3 * D_);
          const int w     = gc - head * 3 * D_;
          const int which = w >> 6;
          const int dd    = w & 63;
          const int bb    = gr / S_;
          const int srow  = gr - bb * S_;
          const size_t dst = ((size_t)(bb * NH_ + head) * S_ + srow) * D_ + dd;
          const unsigned short hv = f32_to_bf16(val);
          if (which == 0)      qO[dst] = hv;
          else if (which == 1) kO[dst] = hv;
          else                 vO[dst] = hv;
        } else if (MODE == 1) {
          outF[(size_t)gr * N + gc] = val + res[(size_t)gr * N + gc];
        } else {
          const float v3 = val * val * val;
          const float gl = 0.5f * val *
              (1.0f + tanhf(0.7978845608028654f * (val + 0.044715f * v3)));
          outH[(size_t)gr * N + gc] = f32_to_bf16(gl);
        }
      }
    }
  }
}

// ---------------------------------------------------------------------------
// Flash attention: one (b,head) per blockIdx.x, 128 queries per block,
// each wave owns 16 queries; 32-key chunks staged cooperatively in LDS
// (K chunk via async copy, V chunk transposed synchronously).
// ---------------------------------------------------------------------------
__global__ __launch_bounds__(256) void attn_wmma(
    const unsigned short* __restrict__ Q,
    const unsigned short* __restrict__ Kb,
    const unsigned short* __restrict__ Vb,
    unsigned short* __restrict__ ctx) {
  __shared__ unsigned short Ks[32][80];       // [key][d]  (B-frags for Q*K^T)
  __shared__ unsigned short Vs[64][40];       // [d][key]  (B-frags for P*V)
  __shared__ unsigned short Ps[8][16][40];    // per-wave P C->A relayout

  const int tid  = threadIdx.x;
  const int lane = tid & 31;
  const int wave = tid >> 5;
  const int rsel = lane & 15;
  const int klo  = (lane >> 4) << 3;
  const int rowoff = klo;

  const int bh   = blockIdx.x;
  const int bb   = bh >> 4;      // NH_ == 16
  const int head = bh & 15;
  const int qbase = blockIdx.y * 128 + wave * 16;

  const unsigned short* qp = Q  + (size_t)bh * S_ * D_;
  const unsigned short* kp = Kb + (size_t)bh * S_ * D_;
  const unsigned short* vp = Vb + (size_t)bh * S_ * D_;

  ABFrag qf[2];
  load_frag16(qf[0], qp,      qbase + rsel, D_, klo);   // K = d 0..31
  load_frag16(qf[1], qp + 32, qbase + rsel, D_, klo);   // K = d 32..63

  CFrag o[4];
  #pragma unroll
  for (int nf = 0; nf < 4; ++nf)
    o[nf].v = (v8f){0.f, 0.f, 0.f, 0.f, 0.f, 0.f, 0.f, 0.f};
  float mI[8], lI[8];
  #pragma unroll
  for (int r = 0; r < 8; ++r) { mI[r] = -1e30f; lI[r] = 0.f; }

  const int krow = tid >> 3, kcol = (tid & 7) << 3;

  for (int kc = 0; kc < S_; kc += 32) {
    // K chunk: straight copy -> async DMA to LDS (16B per thread)
    async_cp16(kp + (size_t)(kc + krow) * D_ + kcol, &Ks[krow][kcol]);
    // V chunk: needs transpose -> synchronous
    U4 uv;
    uv.q = *(const uint4*)(vp + (size_t)(kc + krow) * D_ + kcol);
    #pragma unroll
    for (int j = 0; j < 8; ++j)
      Vs[kcol + j][krow] = uv.h[j];
    wait_async0();
    __syncthreads();

    // scores S = Q * K^T  (two 16-key halves, two K-steps over D=64)
    CFrag sc[2];
    sc[0].v = (v8f){0.f, 0.f, 0.f, 0.f, 0.f, 0.f, 0.f, 0.f};
    sc[1].v = (v8f){0.f, 0.f, 0.f, 0.f, 0.f, 0.f, 0.f, 0.f};
    #pragma unroll
    for (int nhalf = 0; nhalf < 2; ++nhalf) {
      ABFrag kf;
      load_frag16(kf, &Ks[0][0],      nhalf * 16 + rsel, 80, klo);
      sc[nhalf].v = wmma_bf16(qf[0], kf, sc[nhalf].v);
      load_frag16(kf, &Ks[0][0] + 32, nhalf * 16 + rsel, 80, klo);
      sc[nhalf].v = wmma_bf16(qf[1], kf, sc[nhalf].v);
    }

    // online softmax: row reductions across the 16-lane half (wave32)
    #pragma unroll
    for (int r = 0; r < 8; ++r) {
      float s0 = sc[0].f[r] * 0.125f;   // 1/sqrt(64)
      float s1 = sc[1].f[r] * 0.125f;
      float mx = fmaxf(s0, s1);
      #pragma unroll
      for (int off = 1; off < 16; off <<= 1)
        mx = fmaxf(mx, __shfl_xor(mx, off, 32));
      const float mnew = fmaxf(mI[r], mx);
      const float p0 = __expf(s0 - mnew);
      const float p1 = __expf(s1 - mnew);
      float rs = p0 + p1;
      #pragma unroll
      for (int off = 1; off < 16; off <<= 1)
        rs += __shfl_xor(rs, off, 32);
      const float alpha = __expf(mI[r] - mnew);
      lI[r] = lI[r] * alpha + rs;
      mI[r] = mnew;
      #pragma unroll
      for (int nf = 0; nf < 4; ++nf) o[nf].f[r] *= alpha;
      sc[0].f[r] = p0;
      sc[1].f[r] = p1;
    }

    // C-layout P -> A-layout via per-wave LDS scratch (in-order per wave)
    #pragma unroll
    for (int r = 0; r < 8; ++r) {
      Ps[wave][rowoff + r][rsel]      = f32_to_bf16(sc[0].f[r]);
      Ps[wave][rowoff + r][16 + rsel] = f32_to_bf16(sc[1].f[r]);
    }
    ABFrag pf;
    load_frag16(pf, &Ps[wave][0][0], rsel, 40, klo);

    // O += P * V
    #pragma unroll
    for (int nf = 0; nf < 4; ++nf) {
      ABFrag vf;
      load_frag16(vf, &Vs[0][0], nf * 16 + rsel, 40, klo);
      o[nf].v = wmma_bf16(pf, vf, o[nf].v);
    }
    __syncthreads();
  }

  // normalize and scatter ctx (bf16, [B*S, H] with head offset)
  #pragma unroll
  for (int nf = 0; nf < 4; ++nf) {
    const int col = head * D_ + nf * 16 + rsel;
    #pragma unroll
    for (int r = 0; r < 8; ++r) {
      const int qrow = qbase + rowoff + r;
      const float val = o[nf].f[r] / lI[r];
      ctx[((size_t)(bb * S_ + qrow)) * H_ + col] = f32_to_bf16(val);
    }
  }
}

// ---------------------------------------------------------------------------
// Host launcher
// ---------------------------------------------------------------------------
extern "C" void kernel_launch(void* const* d_in, const int* in_sizes, int n_in,
                              void* d_out, int out_size, void* d_ws, size_t ws_size,
                              hipStream_t stream) {
  (void)in_sizes; (void)n_in; (void)out_size; (void)ws_size;
  const float* x      = (const float*)d_in[0];
  const float* ln1_s  = (const float*)d_in[1];
  const float* ln1_b  = (const float*)d_in[2];
  const float* qkv_w  = (const float*)d_in[3];
  const float* qkv_b  = (const float*)d_in[4];
  const float* proj_w = (const float*)d_in[5];
  const float* proj_b = (const float*)d_in[6];
  const float* ln2_s  = (const float*)d_in[7];
  const float* ln2_b  = (const float*)d_in[8];
  const float* wi_w   = (const float*)d_in[9];
  const float* wi_b   = (const float*)d_in[10];
  const float* wo_w   = (const float*)d_in[11];
  const float* wo_b   = (const float*)d_in[12];

  char* ws = (char*)d_ws;
  auto alloc = [&](size_t bytes) -> char* {
    char* p = ws;
    ws += (bytes + 255) & ~(size_t)255;
    return p;
  };

  const size_t nQKVW = (size_t)H_ * 3 * H_;
  const size_t nPROJ = (size_t)H_ * H_;
  const size_t nWI   = (size_t)H_ * FFN_;
  const size_t nWO   = (size_t)FFN_ * H_;
  const size_t nACT  = (size_t)BS_ * H_;
  const size_t nHEAD = (size_t)B_ * NH_ * S_ * D_;
  const size_t nFFN  = (size_t)BS_ * FFN_;

  unsigned short* wqkvT = (unsigned short*)alloc(nQKVW * 2);  // [3H][H]
  unsigned short* wprjT = (unsigned short*)alloc(nPROJ * 2);  // [H][H]
  unsigned short* wwiT  = (unsigned short*)alloc(nWI * 2);    // [FFN][H]
  unsigned short* wwoT  = (unsigned short*)alloc(nWO * 2);    // [H][FFN]
  unsigned short* h1    = (unsigned short*)alloc(nACT * 2);
  unsigned short* qb    = (unsigned short*)alloc(nHEAD * 2);
  unsigned short* kb    = (unsigned short*)alloc(nHEAD * 2);
  unsigned short* vb    = (unsigned short*)alloc(nHEAD * 2);
  unsigned short* ctx   = (unsigned short*)alloc(nACT * 2);
  float*          x2    = (float*)alloc(nACT * 4);
  unsigned short* h2    = (unsigned short*)alloc(nACT * 2);
  unsigned short* ffn   = (unsigned short*)alloc(nFFN * 2);

  // weight conversion fp32 [K][N] -> bf16 [N][K] (transposed for async GEMM)
  cvt_transpose_bf16_kernel<<<dim3(3 * H_ / 32, H_ / 32), 256, 0, stream>>>(
      qkv_w, wqkvT, H_, 3 * H_);
  cvt_transpose_bf16_kernel<<<dim3(H_ / 32, H_ / 32), 256, 0, stream>>>(
      proj_w, wprjT, H_, H_);
  cvt_transpose_bf16_kernel<<<dim3(FFN_ / 32, H_ / 32), 256, 0, stream>>>(
      wi_w, wwiT, H_, FFN_);
  cvt_transpose_bf16_kernel<<<dim3(H_ / 32, FFN_ / 32), 256, 0, stream>>>(
      wo_w, wwoT, FFN_, H_);

  // LN1 -> h1 (bf16)
  ln_bf16_kernel<<<BS_, 256, 0, stream>>>(x, ln1_s, ln1_b, h1);

  // QKV GEMM: [4096,1024] x [1024,3072] -> q/k/v head-major bf16
  gemm_bf16_wmma<0><<<dim3(3 * H_ / 128, BS_ / 128), 256, 0, stream>>>(
      h1, wqkvT, qkv_b, nullptr, nullptr, nullptr, qb, kb, vb, BS_, 3 * H_, H_);

  // flash attention -> ctx bf16
  attn_wmma<<<dim3(B_ * NH_, S_ / 128), 256, 0, stream>>>(qb, kb, vb, ctx);

  // proj + residual(x) -> x2 fp32
  gemm_bf16_wmma<1><<<dim3(H_ / 128, BS_ / 128), 256, 0, stream>>>(
      ctx, wprjT, proj_b, x, x2, nullptr, nullptr, nullptr, nullptr, BS_, H_, H_);

  // LN2 -> h2 (bf16)
  ln_bf16_kernel<<<BS_, 256, 0, stream>>>(x2, ln2_s, ln2_b, h2);

  // FFN up + gelu -> ffn bf16
  gemm_bf16_wmma<2><<<dim3(FFN_ / 128, BS_ / 128), 256, 0, stream>>>(
      h2, wwiT, wi_b, nullptr, nullptr, ffn, nullptr, nullptr, nullptr, BS_, FFN_, H_);

  // FFN down + residual(x2) -> d_out fp32
  gemm_bf16_wmma<1><<<dim3(H_ / 128, BS_ / 128), 256, 0, stream>>>(
      ffn, wwoT, wo_b, x2, (float*)d_out, nullptr, nullptr, nullptr, nullptr, BS_, H_, FFN_);
}
